// TransformerEncoderBlock_25890062860544
// MI455X (gfx1250) — matrix operand order repaired and checked by reading, not compiled, not run on hardware
//
#include <hip/hip_runtime.h>
#include <math.h>

// ---------------------------------------------------------------------------
// Types for CDNA5 WMMA (gfx1250, wave32)
// ---------------------------------------------------------------------------
typedef __bf16 bf16;
typedef __attribute__((ext_vector_type(16))) __bf16 v16bf;
typedef __attribute__((ext_vector_type(8)))  float  v8f;
typedef __attribute__((ext_vector_type(4)))  unsigned int uv4;
typedef __attribute__((ext_vector_type(8)))  unsigned int uv8;

union Frag {           // one WMMA 16-bit A/B operand: 16 bf16 = 32 bytes
    v16bf bf;
    uv4   q[2];
    uv8   o;
};

__device__ __forceinline__ v8f wmma_bf16(v16bf a, v16bf b, v8f c) {
    // D = A(16x32 bf16) * B(32x16 bf16) + C(16x16 f32)
    return __builtin_amdgcn_wmma_f32_16x16x32_bf16(
        /*neg_a=*/false, a, /*neg_b=*/false, b,
        /*c_mod=*/(short)0, c, /*reuse_a=*/false, /*reuse_b=*/false);
}

// CDNA5 async global->LDS copy (16B per lane), tracked by ASYNCcnt.
__device__ __forceinline__ void async_copy16(unsigned lds_off, const void* gptr) {
    asm volatile("global_load_async_to_lds_b128 %0, %1, off"
                 :: "v"(lds_off), "v"(gptr) : "memory");
}
__device__ __forceinline__ void wait_async0() {
    asm volatile("s_wait_asynccnt 0x0" ::: "memory");
}

// 16-lane (row-half) reductions via DPP ladder: xor1, xor2, half-mirror, mirror
template <int CTRL>
__device__ __forceinline__ float dpp_mov_f32(float x) {
    int xi = __builtin_bit_cast(int, x);
    int r  = __builtin_amdgcn_update_dpp(xi, xi, CTRL, 0xF, 0xF, false);
    return __builtin_bit_cast(float, r);
}
__device__ __forceinline__ float red_max16(float x) {
    x = fmaxf(x, dpp_mov_f32<0xB1>(x));   // quad_perm [1,0,3,2]  (xor 1)
    x = fmaxf(x, dpp_mov_f32<0x4E>(x));   // quad_perm [2,3,0,1]  (xor 2)
    x = fmaxf(x, dpp_mov_f32<0x141>(x));  // row_half_mirror      (xor 4 eff.)
    x = fmaxf(x, dpp_mov_f32<0x140>(x));  // row_mirror           (xor 8 eff.)
    return x;
}
__device__ __forceinline__ float red_add16(float x) {
    x += dpp_mov_f32<0xB1>(x);
    x += dpp_mov_f32<0x4E>(x);
    x += dpp_mov_f32<0x141>(x);
    x += dpp_mov_f32<0x140>(x);
    return x;
}

// Problem constants
constexpr int   Bn   = 16;
constexpr int   Nn   = 1024;
constexpr int   En   = 768;
constexpr int   Hn   = 8;
constexpr int   Dn   = 96;           // E / H
constexpr int   HIDn = 1536;         // 2 * E
constexpr int   Mn   = Bn * Nn;      // 16384 rows
constexpr float ATTN_SCALE = 0.036084391824351615f;  // 1/sqrt(E)
constexpr float LN_EPS = 1e-5f;

// ---------------------------------------------------------------------------
// Weight convert + transpose: w[K][N] (f32) -> wT[N][K] (bf16)
// ---------------------------------------------------------------------------
__global__ void __launch_bounds__(256)
cvt_transpose_kernel(const float* __restrict__ w, bf16* __restrict__ wT,
                     int K, int N) {
    int idx = blockIdx.x * 256 + threadIdx.x;
    if (idx < K * N) {
        int k = idx / N;
        int n = idx - k * N;
        wT[(size_t)n * K + k] = (bf16)w[idx];
    }
}

// ---------------------------------------------------------------------------
// LayerNorm over E=768 (one row per 256-thread block) -> bf16
// ---------------------------------------------------------------------------
__global__ void __launch_bounds__(256)
ln_kernel(const float* __restrict__ x, const float* __restrict__ g,
          const float* __restrict__ b, bf16* __restrict__ out) {
    __shared__ float red[16];
    const int row = blockIdx.x;
    const int t   = threadIdx.x;
    const float* xr = x + (size_t)row * En;

    float v0 = xr[t], v1 = xr[t + 256], v2 = xr[t + 512];
    float s  = v0 + v1 + v2;
    float ss = v0 * v0 + v1 * v1 + v2 * v2;
    #pragma unroll
    for (int m = 1; m < 32; m <<= 1) {
        s  += __shfl_xor(s,  m, 32);
        ss += __shfl_xor(ss, m, 32);
    }
    const int w = t >> 5, lane = t & 31;
    if (lane == 0) { red[w] = s; red[8 + w] = ss; }
    __syncthreads();
    if (t == 0) {
        float ts = 0.f, tss = 0.f;
        #pragma unroll
        for (int i = 0; i < 8; i++) { ts += red[i]; tss += red[8 + i]; }
        red[0] = ts; red[8] = tss;
    }
    __syncthreads();
    const float mean = red[0] * (1.0f / 768.0f);
    const float var  = red[8] * (1.0f / 768.0f) - mean * mean;
    const float rstd = rsqrtf(var + LN_EPS);

    bf16* orow = out + (size_t)row * En;
    orow[t]       = (bf16)((v0 - mean) * rstd * g[t]       + b[t]);
    orow[t + 256] = (bf16)((v1 - mean) * rstd * g[t + 256] + b[t + 256]);
    orow[t + 512] = (bf16)((v2 - mean) * rstd * g[t + 512] + b[t + 512]);
}

// ---------------------------------------------------------------------------
// Tiled bf16 WMMA GEMM: C[M,N] = A[M,K] * BT[N,K]^T (+bias, +epilogue)
// Block tile 128x128, 4 waves (2x2), wave tile 64x64 -> 16 WMMA per K-step.
// Double-buffered LDS, staged with CDNA5 async global->LDS DMA
// (global_load_async_to_lds_b128 / s_wait_asynccnt); one barrier per K-step.
//   MODE 0: QKV   -> scatter bf16 to q[b,h,n,d], k[b,h,n,d], vT[b,h,d,n]
//   MODE 1: PROJ  -> f32 out = acc + bias + residual
//   MODE 2: FF1   -> bf16 out = gelu_exact(acc + bias)
//   MODE 3: FF2   -> f32 out = acc + bias + residual
// ---------------------------------------------------------------------------
template <int MODE>
__global__ void __launch_bounds__(128)
gemm_kernel(const bf16* __restrict__ A, const bf16* __restrict__ BT,
            const float* __restrict__ bias, int Kdim, int Nout,
            float* __restrict__ outF, bf16* __restrict__ outB,
            const float* __restrict__ residual,
            bf16* __restrict__ qout, bf16* __restrict__ kout,
            bf16* __restrict__ vtout) {
    __shared__ alignas(64) bf16 As[2][128 * 32];
    __shared__ alignas(64) bf16 Bs[2][128 * 32];

    const int tid  = threadIdx.x;               // 128 threads = 4 waves
    const int lane = tid & 31, wid = tid >> 5;
    const int wm   = wid >> 1, wn = wid & 1;    // 2 x 2 wave grid
    const int kh   = lane >> 4, ln16 = lane & 15;
    const int m0   = blockIdx.y * 128, n0 = blockIdx.x * 128;

    const v8f zero8 = {0.f, 0.f, 0.f, 0.f, 0.f, 0.f, 0.f, 0.f};
    v8f acc[4][4];
    #pragma unroll
    for (int mi = 0; mi < 4; mi++)
        #pragma unroll
        for (int ni = 0; ni < 2 * 2; ni++) acc[mi][ni & 3] = zero8;

    // Each thread stages one 64B row-slice of A tile and of B tile per K-step.
    const bf16* Arow = A  + (size_t)(m0 + tid) * Kdim;
    const bf16* Brow = BT + (size_t)(n0 + tid) * Kdim;
    const unsigned ldsA = (unsigned)(size_t)(&As[0][0]) + (unsigned)tid * 64u;
    const unsigned ldsB = (unsigned)(size_t)(&Bs[0][0]) + (unsigned)tid * 64u;
    const unsigned bufStride = 128u * 32u * 2u;  // bytes per buffer

    auto stage = [&](int buf, int k0) {
        #pragma unroll
        for (int c = 0; c < 4; c++) {
            async_copy16(ldsA + buf * bufStride + c * 16, Arow + k0 + c * 8);
            async_copy16(ldsB + buf * bufStride + c * 16, Brow + k0 + c * 8);
        }
    };

    stage(0, 0);
    wait_async0();
    __syncthreads();

    int buf = 0;
    for (int k0 = 0; k0 < Kdim; k0 += 32, buf ^= 1) {
        const bool has_next = (k0 + 32 < Kdim);
        if (has_next) stage(buf ^ 1, k0 + 32);   // DMA next tile into other buf

        Frag af[4], bfr[4];
        #pragma unroll
        for (int mi = 0; mi < 4; mi++) {
            // A layout: lane holds row M=lane%16, lane-half selects K 0-15 / 16-31
            const int r = wm * 64 + mi * 16 + ln16;
            af[mi].q[0] = *(const uv4*)(&As[buf][r * 32 + kh * 16]);
            af[mi].q[1] = *(const uv4*)(&As[buf][r * 32 + kh * 16 + 8]);
        }
        #pragma unroll
        for (int ni = 0; ni < 4; ni++) {
            // B layout: lane = column n, lane-half selects K 0-15 / 16-31
            const int c = wn * 64 + ni * 16 + ln16;
            bfr[ni].o = *(const uv8*)(&Bs[buf][c * 32 + kh * 16]);
        }
        #pragma unroll
        for (int mi = 0; mi < 4; mi++)
            #pragma unroll
            for (int ni = 0; ni < 4; ni++)
                acc[mi][ni] = wmma_bf16(af[mi].bf, bfr[ni].bf, acc[mi][ni]);

        if (has_next) {
            wait_async0();       // DMA into buf^1 complete
            __syncthreads();     // everyone done reading buf, writes visible
        }
    }

    // Epilogue. C layout: VGPR r -> row (r | r+8 by lane half), col = lane%16
    #pragma unroll
    for (int mi = 0; mi < 4; mi++) {
        #pragma unroll
        for (int ni = 0; ni < 4; ni++) {
            const int colg = n0 + wn * 64 + ni * 16 + ln16;
            const float bv = bias[colg];

            int hidx = 0, dd = 0, tt = 0;
            if (MODE == 0) {                       // col -> (h, d, t) of (H,D,3)
                hidx = colg / (Dn * 3);
                const int rem = colg - hidx * (Dn * 3);
                dd = rem / 3;
                tt = rem - dd * 3;
            }
            #pragma unroll
            for (int r = 0; r < 8; r++) {
                const int rowg = m0 + wm * 64 + mi * 16 + r + kh * 8;
                float v = acc[mi][ni][r] + bv;
                if (MODE == 0) {
                    const int bb = rowg >> 10;     // / N
                    const int nn = rowg & 1023;    // % N
                    const bf16 v16 = (bf16)v;
                    if (tt == 0)
                        qout[(((size_t)(bb * Hn + hidx)) * Nn + nn) * Dn + dd] = v16;
                    else if (tt == 1)
                        kout[(((size_t)(bb * Hn + hidx)) * Nn + nn) * Dn + dd] = v16;
                    else
                        vtout[(((size_t)(bb * Hn + hidx)) * Dn + dd) * Nn + nn] = v16;
                } else if (MODE == 1 || MODE == 3) {
                    outF[(size_t)rowg * Nout + colg] =
                        v + residual[(size_t)rowg * Nout + colg];
                } else { // MODE == 2: exact GELU
                    const float gl = 0.5f * v * (1.0f + erff(v * 0.7071067811865476f));
                    outB[(size_t)rowg * Nout + colg] = (bf16)gl;
                }
            }
        }
    }
}

// ---------------------------------------------------------------------------
// Fused flash attention, one (b,h) column of 64 query rows per 128-thr block.
// Online softmax over 32-key steps; S and PV both via bf16 WMMA; row max/sum
// via DPP ladder (pure VALU, no LDS crossbar traffic).
//   q,k: [B*H, N, D] bf16    vT: [B*H, D, N] bf16    out: [B, N, E] bf16
// ---------------------------------------------------------------------------
__global__ void __launch_bounds__(128)
attn_kernel(const bf16* __restrict__ Q, const bf16* __restrict__ Km,
            const bf16* __restrict__ Vt, bf16* __restrict__ out) {
    __shared__ alignas(32) bf16 Pl[4 * 16 * 32];  // per-wave P re-layout patch

    const int lane = threadIdx.x & 31, w = threadIdx.x >> 5;
    const int kh = lane >> 4, ln16 = lane & 15;
    const int bh  = blockIdx.y;
    const int qr0 = blockIdx.x * 64 + w * 16;

    const bf16* qb = Q  + (size_t)bh * Nn * Dn;
    const bf16* kb = Km + (size_t)bh * Nn * Dn;
    const bf16* vb = Vt + (size_t)bh * Dn * Nn;
    bf16* pl = Pl + w * 512;

    // Q fragments: 16 rows x 96 d  (3 K-steps of 32), loaded once
    Frag qf[3];
    #pragma unroll
    for (int ks = 0; ks < 3; ks++) {
        const bf16* p = qb + (size_t)(qr0 + ln16) * Dn + ks * 32;
        qf[ks].q[0] = *(const uv4*)(p + kh * 16);
        qf[ks].q[1] = *(const uv4*)(p + kh * 16 + 8);
    }

    const v8f zero8 = {0.f, 0.f, 0.f, 0.f, 0.f, 0.f, 0.f, 0.f};
    v8f o_acc[6];
    #pragma unroll
    for (int ni = 0; ni < 6; ni++) o_acc[ni] = zero8;
    float m_run[8], l_run[8];
    #pragma unroll
    for (int r = 0; r < 8; r++) { m_run[r] = -3.0e30f; l_run[r] = 0.f; }

    for (int kt = 0; kt < Nn / 32; kt++) {
        const int key0 = kt * 32;

        // S = Q K^T for two 16-key tiles (B-frag: lane=key col, contiguous d)
        v8f s[2];
        #pragma unroll
        for (int st = 0; st < 2; st++) {
            s[st] = zero8;
            #pragma unroll
            for (int ks = 0; ks < 3; ks++) {
                Frag kf;
                kf.o = *(const uv8*)(kb + (size_t)(key0 + st * 16 + ln16) * Dn
                                        + ks * 32 + kh * 16);
                s[st] = wmma_bf16(qf[ks].bf, kf.bf, s[st]);
            }
        }

        // Online softmax (rows live in 16-lane halves; DPP reductions)
        float p0[8], p1[8], scl[8];
        #pragma unroll
        for (int r = 0; r < 8; r++) {
            const float s0 = s[0][r] * ATTN_SCALE;
            const float s1 = s[1][r] * ATTN_SCALE;
            const float mr = red_max16(fmaxf(s0, s1));
            const float mn = fmaxf(m_run[r], mr);
            const float sc = __expf(m_run[r] - mn);
            const float e0 = __expf(s0 - mn);
            const float e1 = __expf(s1 - mn);
            const float rs = red_add16(e0 + e1);
            l_run[r] = l_run[r] * sc + rs;
            m_run[r] = mn;
            p0[r] = e0; p1[r] = e1; scl[r] = sc;
        }
        #pragma unroll
        for (int ni = 0; ni < 6; ni++)
            #pragma unroll
            for (int r = 0; r < 8; r++) o_acc[ni][r] *= scl[r];

        // C-layout P -> LDS -> A-layout fragment (per-wave region; DS ops from
        // one wave are in-order, no barrier needed)
        #pragma unroll
        for (int r = 0; r < 8; r++) {
            pl[(r + kh * 8) * 32 + ln16]      = (bf16)p0[r];
            pl[(r + kh * 8) * 32 + 16 + ln16] = (bf16)p1[r];
        }
        Frag pf;
        pf.q[0] = *(const uv4*)(pl + ln16 * 32 + kh * 16);
        pf.q[1] = *(const uv4*)(pl + ln16 * 32 + kh * 16 + 8);

        // O += P V   (Vt rows give contiguous per-lane B fragments)
        #pragma unroll
        for (int ni = 0; ni < 6; ni++) {
            Frag vf;
            vf.o = *(const uv8*)(vb + (size_t)(ni * 16 + ln16) * Nn
                                    + key0 + kh * 16);
            o_acc[ni] = wmma_bf16(pf.bf, vf.bf, o_acc[ni]);
        }
    }

    // Normalize and scatter back to [B, N, E] (head-major inside E)
    const int bb = bh >> 3, hh = bh & 7;
    #pragma unroll
    for (int r = 0; r < 8; r++) {
        const float inv = 1.0f / l_run[r];
        const int n = qr0 + r + kh * 8;
        bf16* orow = out + ((size_t)(bb * Nn + n)) * En + hh * Dn;
        #pragma unroll
        for (int ni = 0; ni < 6; ni++)
            orow[ni * 16 + ln16] = (bf16)(o_acc[ni][r] * inv);
    }
}

// ---------------------------------------------------------------------------
// Launch: weights->bf16T, LN1, QKV GEMM, attention, proj(+res), LN2,
//         FF1(+gelu), FF2(+res) -> d_out
// ---------------------------------------------------------------------------
extern "C" void kernel_launch(void* const* d_in, const int* in_sizes, int n_in,
                              void* d_out, int out_size, void* d_ws, size_t ws_size,
                              hipStream_t stream) {
    const float* x      = (const float*)d_in[0];
    const float* ln1_g  = (const float*)d_in[1];
    const float* ln1_b  = (const float*)d_in[2];
    const float* qkv_w  = (const float*)d_in[3];
    const float* qkv_b  = (const float*)d_in[4];
    const float* proj_w = (const float*)d_in[5];
    const float* proj_b = (const float*)d_in[6];
    const float* ln2_g  = (const float*)d_in[7];
    const float* ln2_b  = (const float*)d_in[8];
    const float* ff1_w  = (const float*)d_in[9];
    const float* ff1_b  = (const float*)d_in[10];
    const float* ff2_w  = (const float*)d_in[11];
    const float* ff2_b  = (const float*)d_in[12];
    float* out = (float*)d_out;

    char* ws = (char*)d_ws;
    size_t off = 0;
    auto carve = [&](size_t bytes) -> char* {
        off = (off + 255) & ~(size_t)255;
        char* p = ws + off;
        off += bytes;
        return p;
    };
    const size_t ACT = (size_t)Mn * En * 2;           // 25.2 MB bf16
    bf16* h1   = (bf16*)carve(ACT);
    bf16* qb   = (bf16*)carve(ACT);
    bf16* kbuf = (bf16*)carve(ACT);
    bf16* vT   = (bf16*)carve(ACT);
    bf16* attn = (bf16*)carve(ACT);
    bf16* h2   = (bf16*)carve(ACT);
    bf16* h3   = (bf16*)carve((size_t)Mn * HIDn * 2); // 50.3 MB
    float* x1  = (float*)carve((size_t)Mn * En * 4);  // 50.3 MB
    bf16* wqkvT  = (bf16*)carve((size_t)3 * En * En * 2);
    bf16* wprojT = (bf16*)carve((size_t)En * En * 2);
    bf16* wff1T  = (bf16*)carve((size_t)HIDn * En * 2);
    bf16* wff2T  = (bf16*)carve((size_t)En * HIDn * 2);

    // 1) weights -> bf16 transposed
    cvt_transpose_kernel<<<(En * 3 * En + 255) / 256, 256, 0, stream>>>(qkv_w, wqkvT, En, 3 * En);
    cvt_transpose_kernel<<<(En * En + 255) / 256, 256, 0, stream>>>(proj_w, wprojT, En, En);
    cvt_transpose_kernel<<<(En * HIDn + 255) / 256, 256, 0, stream>>>(ff1_w, wff1T, En, HIDn);
    cvt_transpose_kernel<<<(HIDn * En + 255) / 256, 256, 0, stream>>>(ff2_w, wff2T, HIDn, En);

    // 2) LN1
    ln_kernel<<<Mn, 256, 0, stream>>>(x, ln1_g, ln1_b, h1);

    // 3) QKV GEMM: [16384,768] x [768,2304] -> q/k/vT (bf16)
    gemm_kernel<0><<<dim3(3 * En / 128, Mn / 128), 128, 0, stream>>>(
        h1, wqkvT, qkv_b, En, 3 * En, nullptr, nullptr, nullptr, qb, kbuf, vT);

    // 4) fused flash attention -> attn [B,N,E] bf16
    attn_kernel<<<dim3(Nn / 64, Bn * Hn), 128, 0, stream>>>(qb, kbuf, vT, attn);

    // 5) proj GEMM + bias + residual(x) -> x1 (f32)
    gemm_kernel<1><<<dim3(En / 128, Mn / 128), 128, 0, stream>>>(
        attn, wprojT, proj_b, En, En, x1, nullptr, x, nullptr, nullptr, nullptr);

    // 6) LN2
    ln_kernel<<<Mn, 256, 0, stream>>>(x1, ln2_g, ln2_b, h2);

    // 7) FF1 GEMM + bias + exact GELU -> h3 (bf16)
    gemm_kernel<2><<<dim3(HIDn / 128, Mn / 128), 128, 0, stream>>>(
        h2, wff1T, ff1_b, En, HIDn, nullptr, h3, nullptr, nullptr, nullptr, nullptr);

    // 8) FF2 GEMM + bias + residual(x1) -> d_out (f32)
    gemm_kernel<3><<<dim3(En / 128, Mn / 128), 128, 0, stream>>>(
        h3, wff2T, ff2_b, HIDn, En, out, nullptr, x1, nullptr, nullptr, nullptr);
}
